// GroupedQueryAttention_57724360458753
// MI455X (gfx1250) — matrix operand order repaired and checked by reading, compiled, not run on hardware
//
#include <hip/hip_runtime.h>

// ---------------- problem constants ----------------
#define BATCH 2
#define S_LEN 2048
#define HID   2560
#define NH    32
#define NKV   8
#define HD    80      // head dim
#define HDP   96      // padded head dim (3 x 32 for bf16 WMMA K-steps)
#define KVD   (NKV * HD)   // 640

// ---------------- WMMA types ----------------
typedef __attribute__((ext_vector_type(16))) __bf16  v16bf;
typedef __attribute__((ext_vector_type(8)))  float   v8f;
typedef __attribute__((ext_vector_type(4)))  unsigned u32x4;

union Frag16 { u32x4 u[2]; v16bf v; };

__device__ __forceinline__ unsigned short f2bf(float f) {
  union { float f; unsigned u; } x; x.f = f;
  unsigned r = x.u + 0x7FFFu + ((x.u >> 16) & 1u);   // round-to-nearest-even
  return (unsigned short)(r >> 16);
}

__device__ __forceinline__ v8f wmma_bf16(const Frag16& a, const Frag16& b, v8f c) {
  return __builtin_amdgcn_wmma_f32_16x16x32_bf16(false, a.v, false, b.v,
                                                 (short)0, c, false, false);
}

// ---- CDNA5 async global->LDS copy (ASYNCcnt path), 16B per lane ----
__device__ __forceinline__ void async_copy_b128(void* lds_dst, const void* gsrc) {
  unsigned loff = (unsigned)(unsigned long long)lds_dst;  // low 32 bits of the pointer are the LDS address
  asm volatile("global_load_async_to_lds_b128 %0, %1, off"
               :: "v"(loff), "v"(gsrc) : "memory");
}
__device__ __forceinline__ void wait_async() {
#if __has_builtin(__builtin_amdgcn_s_wait_asynccnt)
  __builtin_amdgcn_s_wait_asynccnt(0);
#else
  asm volatile("s_wait_asynccnt 0x0" ::: "memory");
#endif
}

// =====================================================================
// One-time precision/layout prepasses (memory-bound, tiny vs GEMM reuse)
// =====================================================================
__global__ __launch_bounds__(256)
void cvt_bf16_kernel(const float* __restrict__ s, unsigned short* __restrict__ d) {
  const long i = (long)blockIdx.x * 256 + threadIdx.x;
  d[i] = f2bf(s[i]);
}
// fp32 [R][C] -> bf16 [C][R]
__global__ __launch_bounds__(256)
void cvt_t_bf16_kernel(const float* __restrict__ s, unsigned short* __restrict__ d,
                       int R, int C) {
  const long i = (long)blockIdx.x * 256 + threadIdx.x;
  const int r = (int)(i / C), c = (int)(i % C);
  d[(long)c * R + r] = f2bf(s[i]);
}

// =====================================================================
// bf16 WMMA GEMM:  C[M,N] = A[M,K] * WT[N,K]^T   (N, K compile-time)
// 256 threads = 8 waves; 128x64 tile; BK=32; wave computes 32x32 (4 WMMAs).
// Double-buffered LDS; next tile async-staged while current is computed.
// =====================================================================
template <int N, int K>
__global__ __launch_bounds__(256)
void gemm_bf16wmma(const unsigned short* __restrict__ A,
                   const unsigned short* __restrict__ WT,
                   float* __restrict__ C, int M) {
  __shared__ alignas(16) unsigned short As[2][128][32];  // 16 KB
  __shared__ alignas(16) unsigned short Wt[2][64][32];   //  8 KB

  const int tid  = threadIdx.x;
  const int lane = tid & 31;
  const int w    = tid >> 5;
  const int wr   = w >> 1;       // 0..3 -> 32-row band
  const int wc   = w & 1;        // 0..1 -> 32-col band
  const int ml   = lane & 15;
  const int half = lane >> 4;

  const int m0 = blockIdx.y * 128;
  const int n0 = blockIdx.x * 64;

  // branch-free staging assignment (A: 512 b128 units, W: 256 units)
  const int arow0 = tid >> 2,         apart = tid & 3;
  const int arow1 = (tid + 256) >> 2;
  const int brow  = tid >> 2;

  const unsigned short* Abase = A  + (long)m0 * K;
  const unsigned short* Bbase = WT + (long)n0 * K;

  auto stage = [&](int buf, int k0) {
    async_copy_b128(&As[buf][arow0][apart * 8], Abase + (long)arow0 * K + k0 + apart * 8);
    async_copy_b128(&As[buf][arow1][apart * 8], Abase + (long)arow1 * K + k0 + apart * 8);
    async_copy_b128(&Wt[buf][brow][apart * 8],  Bbase + (long)brow  * K + k0 + apart * 8);
  };

  v8f a00 = {0.f,0.f,0.f,0.f,0.f,0.f,0.f,0.f}, a01 = a00, a10 = a00, a11 = a00;

  stage(0, 0);
  wait_async();
  __syncthreads();

  for (int k0 = 0; k0 < K; k0 += 32) {
    const int cur = (k0 >> 5) & 1;
    if (k0 + 32 < K) stage(cur ^ 1, k0 + 32);   // prefetch next tile (no wait)

    Frag16 fa0, fa1, fb0, fb1;
    const unsigned short* ap0 = &As[cur][32 * wr + ml][0];
    fa0.u[0] = *(const u32x4*)(ap0 + half * 8);
    fa0.u[1] = *(const u32x4*)(ap0 + 16 + half * 8);
    const unsigned short* ap1 = &As[cur][32 * wr + 16 + ml][0];
    fa1.u[0] = *(const u32x4*)(ap1 + half * 8);
    fa1.u[1] = *(const u32x4*)(ap1 + 16 + half * 8);
    const unsigned short* bp0 = &Wt[cur][32 * wc + ml][half * 8];
    fb0.u[0] = *(const u32x4*)bp0;  fb0.u[1] = *(const u32x4*)(bp0 + 16);
    const unsigned short* bp1 = &Wt[cur][32 * wc + 16 + ml][half * 8];
    fb1.u[0] = *(const u32x4*)bp1;  fb1.u[1] = *(const u32x4*)(bp1 + 16);

    a00 = wmma_bf16(fa0, fb0, a00);
    a01 = wmma_bf16(fa0, fb1, a01);
    a10 = wmma_bf16(fa1, fb0, a10);
    a11 = wmma_bf16(fa1, fb1, a11);

    wait_async();        // next buffer landed (this wave)
    __syncthreads();     // all waves done reading cur & done staging
  }

  const int nb = n0 + 32 * wc + ml;
#pragma unroll
  for (int j = 0; j < 8; ++j) {
    const long mg0 = m0 + 32 * wr + half * 8 + j;
    const long mg1 = mg0 + 16;
    C[mg0 * N + nb]      = a00[j];
    C[mg0 * N + nb + 16] = a01[j];
    C[mg1 * N + nb]      = a10[j];
    C[mg1 * N + nb + 16] = a11[j];
  }
}

// =====================================================================
// RoPE + layout pack:
//   Q -> qpad  [B, NH, S, 96]  bf16 (pad cols pre-zeroed by memset)
//   K -> kpad  [B, NKV, S, 96] bf16
//   V -> vt    [B, NKV, 80, S] bf16 (transposed for P*V B-fragments)
// =====================================================================
__global__ __launch_bounds__(256)
void rope_pack_kernel(const float* __restrict__ qbuf, const float* __restrict__ kbuf,
                      const float* __restrict__ vbuf, const float* __restrict__ cosp,
                      const float* __restrict__ sinp, unsigned short* __restrict__ qpad,
                      unsigned short* __restrict__ kpad, unsigned short* __restrict__ vt) {
  const long idx = (long)blockIdx.x * blockDim.x + threadIdx.x;
  const int col = (int)(idx % 3840);
  const long row = idx / 3840;            // b*S + s
  const int s = (int)(row % S_LEN);
  const int b = (int)(row / S_LEN);

  if (col < HID) {                        // ---- Q + RoPE
    const int h = col / HD, d = col % HD;
    const float x = qbuf[row * HID + col];
    const int pd = (d < 40) ? d + 40 : d - 40;
    const float p = qbuf[row * HID + h * HD + pd];
    const float r = (d < 40) ? -p : p;
    const float val = x * cosp[s * HD + d] + r * sinp[s * HD + d];
    qpad[((long)(b * NH + h) * S_LEN + s) * HDP + d] = f2bf(val);
  } else if (col < HID + KVD) {           // ---- K + RoPE
    const int c2 = col - HID;
    const int h = c2 / HD, d = c2 % HD;
    const float x = kbuf[row * KVD + c2];
    const int pd = (d < 40) ? d + 40 : d - 40;
    const float p = kbuf[row * KVD + h * HD + pd];
    const float r = (d < 40) ? -p : p;
    const float val = x * cosp[s * HD + d] + r * sinp[s * HD + d];
    kpad[((long)(b * NKV + h) * S_LEN + s) * HDP + d] = f2bf(val);
  } else {                                // ---- V transpose
    const int c2 = col - (HID + KVD);
    const int h = c2 / HD, d = c2 % HD;
    vt[((long)(b * NKV + h) * HD + d) * S_LEN + s] = f2bf(vbuf[row * KVD + c2]);
  }
}

// =====================================================================
// Flash attention, causal, GQA (G=4). Block = (qblk, h, b), 128 thr = 4 waves.
// Wave w owns 16 query rows; kv tiles of 32, double-buffered async staging.
// Writes ctx directly in bf16 for the output projection.
// =====================================================================
__global__ __launch_bounds__(128)
void attn_kernel(const unsigned short* __restrict__ qpad,
                 const unsigned short* __restrict__ kpad,
                 const unsigned short* __restrict__ vt,
                 unsigned short* __restrict__ ctx) {
  __shared__ alignas(16) unsigned short Klds[2][32][96];   // 12 KB
  __shared__ alignas(16) unsigned short Vlds[2][80][32];   // 10 KB
  __shared__ alignas(16) unsigned short Plds[4][16][32];   //  4 KB

  const int qblk = blockIdx.x, h = blockIdx.y, b = blockIdx.z;
  const int kvh  = h >> 2;                     // G = 4
  const int tid  = threadIdx.x;
  const int lane = tid & 31, w = tid >> 5;
  const int ml   = lane & 15, half = lane >> 4;
  const int qbase = qblk * 64;

  // ---- preload Q fragments (3 K-steps of 32 over padded D=96)
  Frag16 aq[3];
  {
    const unsigned short* qp =
        qpad + ((long)(b * NH + h) * S_LEN + qbase + 16 * w + ml) * HDP;
#pragma unroll
    for (int ks = 0; ks < 3; ++ks) {
      aq[ks].u[0] = *(const u32x4*)(qp + ks * 32 + half * 8);
      aq[ks].u[1] = *(const u32x4*)(qp + ks * 32 + 16 + half * 8);
    }
  }

  float row_max[8], row_sum[8];
  v8f o[5];
#pragma unroll
  for (int j = 0; j < 8; ++j) { row_max[j] = -1e30f; row_sum[j] = 0.f; }
#pragma unroll
  for (int d2 = 0; d2 < 5; ++d2) o[d2] = (v8f){0.f,0.f,0.f,0.f,0.f,0.f,0.f,0.f};

  const unsigned short* kbase_ptr = kpad + (long)(b * NKV + kvh) * S_LEN * HDP;
  const unsigned short* vbase_ptr = vt   + (long)(b * NKV + kvh) * HD * S_LEN;
  const int ntiles = (qbase + 64) >> 5;        // causal bound
  const float scale = 0.11180339887498949f;    // 1/sqrt(80)

  // V staging assignment: 320 b128 units (unit u: d=u>>2, part=u&3)
  const int vd0 = tid >> 2, vp = tid & 3;
  const int vd1 = (tid + 128) >> 2;
  const int vd2 = (tid + 256) >> 2;

  auto stage = [&](int buf, int kv0) {
    const char* ksrc = (const char*)(kbase_ptr + (long)kv0 * HDP);
    char* kdst = (char*)&Klds[buf][0][0];
    async_copy_b128(kdst + tid * 16,         ksrc + tid * 16);
    async_copy_b128(kdst + (tid + 128) * 16, ksrc + (tid + 128) * 16);
    async_copy_b128(kdst + (tid + 256) * 16, ksrc + (tid + 256) * 16);
    char* vdst = (char*)&Vlds[buf][0][0];
    async_copy_b128(vdst + tid * 16,
                    vbase_ptr + (long)vd0 * S_LEN + kv0 + vp * 8);
    async_copy_b128(vdst + (tid + 128) * 16,
                    vbase_ptr + (long)vd1 * S_LEN + kv0 + vp * 8);
    if (tid < 64)
      async_copy_b128(vdst + (tid + 256) * 16,
                      vbase_ptr + (long)vd2 * S_LEN + kv0 + vp * 8);
  };

  stage(0, 0);
  wait_async();
  __syncthreads();

  for (int t = 0; t < ntiles; ++t) {
    const int kv0 = t * 32;
    const int cur = t & 1;
    if (t + 1 < ntiles) stage(cur ^ 1, kv0 + 32);   // prefetch next kv tile

    // ---- scores: S = Q * K^T  (2 col-subtiles x 3 K-steps)
    v8f sc0 = {0.f,0.f,0.f,0.f,0.f,0.f,0.f,0.f};
    v8f sc1 = sc0;
#pragma unroll
    for (int ks = 0; ks < 3; ++ks) {
      Frag16 bk0, bk1;
      const unsigned short* kp0 = &Klds[cur][ml][ks * 32 + half * 8];
      bk0.u[0] = *(const u32x4*)kp0;  bk0.u[1] = *(const u32x4*)(kp0 + 16);
      const unsigned short* kp1 = &Klds[cur][16 + ml][ks * 32 + half * 8];
      bk1.u[0] = *(const u32x4*)kp1;  bk1.u[1] = *(const u32x4*)(kp1 + 16);
      sc0 = wmma_bf16(aq[ks], bk0, sc0);
      sc1 = wmma_bf16(aq[ks], bk1, sc1);
    }

    // ---- causal mask + online softmax (rows live across one 16-lane half)
#pragma unroll
    for (int j = 0; j < 8; ++j) {
      const int mg = qbase + 16 * w + half * 8 + j;
      float s0 = sc0[j] * scale;
      float s1 = sc1[j] * scale;
      if (kv0 + ml > mg)      s0 = -1e30f;
      if (kv0 + 16 + ml > mg) s1 = -1e30f;
      float mx = fmaxf(s0, s1);
#pragma unroll
      for (int off = 1; off < 16; off <<= 1) mx = fmaxf(mx, __shfl_xor(mx, off, 32));
      const float mnew  = fmaxf(row_max[j], mx);
      const float alpha = __expf(row_max[j] - mnew);
      const float p0 = __expf(s0 - mnew);
      const float p1 = __expf(s1 - mnew);
      float ps = p0 + p1;
#pragma unroll
      for (int off = 1; off < 16; off <<= 1) ps += __shfl_xor(ps, off, 32);
      row_sum[j] = row_sum[j] * alpha + ps;
      row_max[j] = mnew;
#pragma unroll
      for (int d2 = 0; d2 < 5; ++d2) o[d2][j] *= alpha;
      Plds[w][half * 8 + j][ml]      = f2bf(p0);
      Plds[w][half * 8 + j][16 + ml] = f2bf(p1);
    }

    // ---- O += P * V   (1 K-step of 32 kv, 5 d-subtiles)
    Frag16 ap;
    const unsigned short* pp = &Plds[w][ml][0];
    ap.u[0] = *(const u32x4*)(pp + half * 8);
    ap.u[1] = *(const u32x4*)(pp + 16 + half * 8);
#pragma unroll
    for (int d2 = 0; d2 < 5; ++d2) {
      Frag16 bv;
      const unsigned short* vp2 = &Vlds[cur][d2 * 16 + ml][half * 8];
      bv.u[0] = *(const u32x4*)vp2;  bv.u[1] = *(const u32x4*)(vp2 + 16);
      o[d2] = wmma_bf16(ap, bv, o[d2]);
    }

    wait_async();        // next kv tile landed (this wave)
    __syncthreads();     // all waves done reading cur & done staging
  }

  // ---- epilogue: normalize, write bf16 ctx[b*S+s][h*80+d]
#pragma unroll
  for (int j = 0; j < 8; ++j) {
    const int sg = qbase + 16 * w + half * 8 + j;
    const float inv = 1.0f / row_sum[j];
    const long rowoff = ((long)(b * S_LEN + sg)) * HID + h * HD;
#pragma unroll
    for (int d2 = 0; d2 < 5; ++d2)
      ctx[rowoff + d2 * 16 + ml] = f2bf(o[d2][j] * inv);
  }
}

// =====================================================================
extern "C" void kernel_launch(void* const* d_in, const int* in_sizes, int n_in,
                              void* d_out, int out_size, void* d_ws, size_t ws_size,
                              hipStream_t stream) {
  (void)in_sizes; (void)n_in; (void)out_size; (void)ws_size;
  const float* hidden = (const float*)d_in[0];
  const float* cosp   = (const float*)d_in[1];
  const float* sinp   = (const float*)d_in[2];
  const float* Wq     = (const float*)d_in[3];
  const float* Wk     = (const float*)d_in[4];
  const float* Wv     = (const float*)d_in[5];
  const float* Wo     = (const float*)d_in[6];
  float* out = (float*)d_out;

  const int M = BATCH * S_LEN;                      // 4096
  char* ws = (char*)d_ws;
  size_t off = 0;
  unsigned short* hidb = (unsigned short*)(ws + off); off += (size_t)M * HID * 2;
  unsigned short* wqT  = (unsigned short*)(ws + off); off += (size_t)HID * HID * 2;
  unsigned short* wkT  = (unsigned short*)(ws + off); off += (size_t)HID * KVD * 2;
  unsigned short* wvT  = (unsigned short*)(ws + off); off += (size_t)HID * KVD * 2;
  unsigned short* woT  = (unsigned short*)(ws + off); off += (size_t)HID * HID * 2;
  float* qbuf          = (float*)(ws + off);          off += (size_t)M * HID * 4;
  float* kbuf          = (float*)(ws + off);          off += (size_t)M * KVD * 4;
  float* vbuf          = (float*)(ws + off);          off += (size_t)M * KVD * 4;
  unsigned short* qpad = (unsigned short*)(ws + off);
  const size_t SZ_QPAD = (size_t)BATCH * NH  * S_LEN * HDP * 2; off += SZ_QPAD;
  unsigned short* kpad = (unsigned short*)(ws + off);
  const size_t SZ_KPAD = (size_t)BATCH * NKV * S_LEN * HDP * 2; off += SZ_KPAD;
  unsigned short* vtp  = (unsigned short*)(ws + off);
  unsigned short* ctx  = (unsigned short*)qbuf;   // qbuf dead after rope_pack

  // zero pad columns of qpad/kpad (contiguous regions)
  hipMemsetAsync(qpad, 0, SZ_QPAD + SZ_KPAD, stream);

  // one-time bf16 conversion / weight transposition
  cvt_bf16_kernel<<<(unsigned)((size_t)M * HID / 256), 256, 0, stream>>>(hidden, hidb);
  cvt_t_bf16_kernel<<<(unsigned)((size_t)HID * HID / 256), 256, 0, stream>>>(Wq, wqT, HID, HID);
  cvt_t_bf16_kernel<<<(unsigned)((size_t)HID * KVD / 256), 256, 0, stream>>>(Wk, wkT, HID, KVD);
  cvt_t_bf16_kernel<<<(unsigned)((size_t)HID * KVD / 256), 256, 0, stream>>>(Wv, wvT, HID, KVD);
  cvt_t_bf16_kernel<<<(unsigned)((size_t)HID * HID / 256), 256, 0, stream>>>(Wo, woT, HID, HID);

  // QKV projections (bf16 WMMA GEMMs, double-buffered async LDS staging)
  gemm_bf16wmma<HID, HID><<<dim3(HID / 64, M / 128), 256, 0, stream>>>(hidb, wqT, qbuf, M);
  gemm_bf16wmma<KVD, HID><<<dim3(KVD / 64, M / 128), 256, 0, stream>>>(hidb, wkT, kbuf, M);
  gemm_bf16wmma<KVD, HID><<<dim3(KVD / 64, M / 128), 256, 0, stream>>>(hidb, wvT, vbuf, M);

  // RoPE + pack to attention-friendly bf16 layouts
  const long total = (long)M * 3840;
  rope_pack_kernel<<<(unsigned)(total / 256), 256, 0, stream>>>(
      qbuf, kbuf, vbuf, cosp, sinp, qpad, kpad, vtp);

  // causal GQA flash attention (bf16 ctx out)
  attn_kernel<<<dim3(S_LEN / 64, NH, BATCH), 128, 0, stream>>>(qpad, kpad, vtp, ctx);

  // output projection
  gemm_bf16wmma<HID, HID><<<dim3(HID / 64, M / 128), 256, 0, stream>>>(ctx, woT, out, M);
}
